// ClassifierB_14267881357432
// MI455X (gfx1250) — matrix operand order repaired and checked by reading, not compiled, not run on hardware
//
#include <hip/hip_runtime.h>
#include <hip/hip_bf16.h>

// ---------------- types for WMMA ----------------
typedef __attribute__((ext_vector_type(16))) __bf16 v16bf;
typedef __attribute__((ext_vector_type(8)))  float  v8f;

union Frag16 { v16bf v; unsigned int u[8]; };

__device__ __forceinline__ unsigned short f2bf(float f) {
    unsigned int u = __float_as_uint(f);
    unsigned int r = u + 0x7FFFu + ((u >> 16) & 1u);   // RNE
    return (unsigned short)(r >> 16);
}

// ---------------- elementwise kernels ----------------
__global__ void k_cvt(const float* __restrict__ in, unsigned short* __restrict__ out, long long n) {
    long long i = (long long)blockIdx.x * blockDim.x + threadIdx.x;
    if (i < n) out[i] = f2bf(in[i]);
}

__global__ void k_zero(float* __restrict__ p, long long n) {
    long long i = (long long)blockIdx.x * blockDim.x + threadIdx.x;
    if (i < n) p[i] = 0.0f;
}

// ---------------- generic WMMA GEMM ----------------
// C[M,N] = act( A[M,K](bf16) * B + bias ),  accumulate f32.
// bWT=1: Bm is W[N,K] row-major (Linear, uses W[n*ldb+k])
// bWT=0: Bm is B[K,N] row-major (uses B[k*ldb+n])
// batching over blockIdx.z with element strides sA/sB/sC. act: 0 none, 1 relu, 2 tanh.
#define TBM 128
#define TBN 128
#define TBK 32
#define LDSROW 20   // 16 data dwords + 4 pad (80B rows, 16B aligned)

__global__ __launch_bounds__(256) void k_gemm(
    const unsigned short* __restrict__ A, long long sA, int lda,
    const unsigned short* __restrict__ Bm, long long sB, int ldb, int bWT,
    const float* __restrict__ bias,
    float* __restrict__ C, long long sC,
    unsigned short* __restrict__ Cb, long long sCb,
    int M, int N, int K, int act, int ldc)
{
    __shared__ unsigned int lA[2][TBM * LDSROW];
    __shared__ unsigned int lB[2][TBN * LDSROW];

    const int tid  = threadIdx.x;
    const int wid  = tid >> 5;
    const int lane = tid & 31;
    const int wm   = wid & 1;        // 2 waves over M (64 rows each)
    const int wn   = wid >> 1;       // 4 waves over N (32 cols each)
    const int l15  = lane & 15;
    const int half = lane >> 4;

    const int z  = blockIdx.z;
    const int bm = blockIdx.y * TBM;
    const int bn = blockIdx.x * TBN;

    A  += (long long)z * sA;
    Bm += (long long)z * sB;

    const bool fullM = (bm + TBM <= M);
    const bool fullN = (bn + TBN <= N);

    // ---- staging helpers (b128 global loads -> b128 LDS stores) ----
    auto stageA = [&](unsigned int* dst, int kt) {
        if (fullM) {
            #pragma unroll
            for (int j = 0; j < 2; ++j) {
                int idx = j * 256 + tid;            // 512 x uint4 chunks
                int r = idx >> 2, c4 = idx & 3;
                uint4 v = *(const uint4*)(A + (long long)(bm + r) * lda + kt + c4 * 8);
                *(uint4*)&dst[r * LDSROW + c4 * 4] = v;
            }
        } else {
            #pragma unroll
            for (int j = 0; j < 2; ++j) {
                int idx = j * 256 + tid;
                int r = idx >> 2, c4 = idx & 3;
                uint4 v = {0u, 0u, 0u, 0u};
                if (bm + r < M) v = *(const uint4*)(A + (long long)(bm + r) * lda + kt + c4 * 8);
                *(uint4*)&dst[r * LDSROW + c4 * 4] = v;
            }
        }
    };
    auto stageB = [&](unsigned int* dst, int kt) {
        if (bWT) {
            if (fullN) {
                #pragma unroll
                for (int j = 0; j < 2; ++j) {
                    int idx = j * 256 + tid;
                    int r = idx >> 2, c4 = idx & 3;
                    uint4 v = *(const uint4*)(Bm + (long long)(bn + r) * ldb + kt + c4 * 8);
                    *(uint4*)&dst[r * LDSROW + c4 * 4] = v;
                }
            } else {
                #pragma unroll
                for (int j = 0; j < 2; ++j) {
                    int idx = j * 256 + tid;
                    int r = idx >> 2, c4 = idx & 3;
                    uint4 v = {0u, 0u, 0u, 0u};
                    if (bn + r < N) v = *(const uint4*)(Bm + (long long)(bn + r) * ldb + kt + c4 * 8);
                    *(uint4*)&dst[r * LDSROW + c4 * 4] = v;
                }
            }
        } else {
            // transpose gather: Bt[n][k] = Bm[k][n]; read 2 n's per b32
            unsigned short* dh = (unsigned short*)dst;
            if (fullN) {
                #pragma unroll
                for (int j = 0; j < 8; ++j) {
                    int idx = j * 256 + tid;        // 2048 x u32 (pairs over n)
                    int n2 = idx & 63, k = idx >> 6;
                    unsigned int v = *(const unsigned int*)(Bm + (long long)(kt + k) * ldb + bn + n2 * 2);
                    dh[(n2 * 2)     * (LDSROW * 2) + k] = (unsigned short)(v & 0xFFFFu);
                    dh[(n2 * 2 + 1) * (LDSROW * 2) + k] = (unsigned short)(v >> 16);
                }
            } else {
                #pragma unroll
                for (int j = 0; j < 16; ++j) {
                    int idx = j * 256 + tid;        // 4096 x u16
                    int n = idx & 127, k = idx >> 7;
                    unsigned short v = 0;
                    if (bn + n < N) v = Bm[(long long)(kt + k) * ldb + bn + n];
                    dh[n * (LDSROW * 2) + k] = v;
                }
            }
        }
    };

    v8f acc[4][2] = {};

    int cur = 0;
    stageA(lA[0], 0);
    stageB(lB[0], 0);
    __syncthreads();

    for (int kt = 0; kt < K; kt += TBK) {
        int nxt = cur ^ 1;
        if (kt + TBK < K) {             // prefetch next tile into other buffer
            stageA(lA[nxt], kt + TBK);
            stageB(lB[nxt], kt + TBK);
        }

        // ---- fragments from current buffer ----
        Frag16 af[4], bfm[2];
        #pragma unroll
        for (int mi = 0; mi < 4; ++mi) {
            const unsigned int* rp = &lA[cur][(wm * 64 + mi * 16 + l15) * LDSROW + half * 8];
            #pragma unroll
            for (int q = 0; q < 8; ++q) af[mi].u[q] = rp[q];
        }
        #pragma unroll
        for (int ni = 0; ni < 2; ++ni) {
            const unsigned int* rp = &lB[cur][(wn * 32 + ni * 16 + l15) * LDSROW + half * 8];
            #pragma unroll
            for (int q = 0; q < 8; ++q) bfm[ni].u[q] = rp[q];
        }
        #pragma unroll
        for (int mi = 0; mi < 4; ++mi)
            #pragma unroll
            for (int ni = 0; ni < 2; ++ni)
                acc[mi][ni] = __builtin_amdgcn_wmma_f32_16x16x32_bf16(
                    false, af[mi].v, false, bfm[ni].v, (short)0, acc[mi][ni], false, false);

        __syncthreads();                // next-buffer stores done; current safe to reuse
        cur = nxt;
    }

    // ---- epilogue: C/D layout: lane&15 = N col, VGPR r + 8*(lane>>4) = M row ----
    if (C)  C  += (long long)z * sC;
    if (Cb) Cb += (long long)z * sCb;
    if (fullM && fullN) {
        #pragma unroll
        for (int mi = 0; mi < 4; ++mi) {
            #pragma unroll
            for (int ni = 0; ni < 2; ++ni) {
                int nn = bn + wn * 32 + ni * 16 + l15;
                float bv = bias ? bias[nn] : 0.0f;
                #pragma unroll
                for (int r = 0; r < 8; ++r) {
                    int mm = bm + wm * 64 + mi * 16 + half * 8 + r;
                    float v = acc[mi][ni][r] + bv;
                    if (act == 1) v = fmaxf(v, 0.0f);
                    else if (act == 2) v = tanhf(v);
                    if (C)  C[(long long)mm * ldc + nn] = v;
                    if (Cb) Cb[(long long)mm * ldc + nn] = f2bf(v);
                }
            }
        }
    } else {
        #pragma unroll
        for (int mi = 0; mi < 4; ++mi) {
            #pragma unroll
            for (int ni = 0; ni < 2; ++ni) {
                int nn = bn + wn * 32 + ni * 16 + l15;
                #pragma unroll
                for (int r = 0; r < 8; ++r) {
                    int mm = bm + wm * 64 + mi * 16 + half * 8 + r;
                    if (mm < M && nn < N) {
                        float v = acc[mi][ni][r];
                        if (bias) v += bias[nn];
                        if (act == 1) v = fmaxf(v, 0.0f);
                        else if (act == 2) v = tanhf(v);
                        if (C)  C[(long long)mm * ldc + nn] = v;
                        if (Cb) Cb[(long long)mm * ldc + nn] = f2bf(v);
                    }
                }
            }
        }
    }
}

// ---------------- softmax over rows (dim -2), writes transposed C^T in bf16 ----------------
// f: [16][384][384] f32, ct: [16][384][384] bf16 with ct[b][m][n] = softmax_n'(f[b][:, m])[n]
__global__ void k_softmax_ct(const float* __restrict__ f, unsigned short* __restrict__ ct) {
    int id = blockIdx.x * blockDim.x + threadIdx.x;    // 16*384 columns
    if (id >= 16 * 384) return;
    int b = id / 384, m = id % 384;
    const float* col = f + (long long)b * 384 * 384 + m;
    float mx = -3.4e38f;
    for (int n = 0; n < 384; ++n) mx = fmaxf(mx, col[(long long)n * 384]);
    float s = 0.0f;
    for (int n = 0; n < 384; ++n) s += expf(col[(long long)n * 384] - mx);
    float inv = 1.0f / s;
    unsigned short* out = ct + (long long)b * 384 * 384 + (long long)m * 384;
    for (int n = 0; n < 384; ++n)
        out[n] = f2bf(expf(col[(long long)n * 384] - mx) * inv);
}

// ---------------- BatchNorm helpers ----------------
// st: [0..1023]=sum, [1024..2047]=sumsq. Each block reduces 64 rows of 1024 channels.
__global__ void k_bn_stats(const float* __restrict__ x, float* __restrict__ st, int rows) {
    int r0 = blockIdx.x * 64;
    for (int c = threadIdx.x; c < 1024; c += 256) {
        float s = 0.0f, s2 = 0.0f;
        for (int r = 0; r < 64; ++r) {
            float v = x[(long long)(r0 + r) * 1024 + c];
            s += v; s2 += v * v;
        }
        atomicAdd(&st[c], s);
        atomicAdd(&st[1024 + c], s2);
    }
}

__global__ void k_bn_coeff(const float* __restrict__ st, const float* __restrict__ g,
                           const float* __restrict__ be, float* __restrict__ cf, float inv) {
    int c = blockIdx.x * blockDim.x + threadIdx.x;
    if (c < 1024) {
        float mu  = st[c] * inv;
        float var = st[1024 + c] * inv - mu * mu;
        float sc  = g[c] * rsqrtf(var + 1e-5f);
        cf[c] = sc;
        cf[1024 + c] = be[c] - mu * sc;
    }
}

__global__ void k_bn1_apply(const float* __restrict__ x, const float* __restrict__ cf,
                            unsigned short* __restrict__ y, long long n) {
    long long i = (long long)blockIdx.x * blockDim.x + threadIdx.x;
    if (i >= n) return;
    int c = (int)(i & 1023);
    y[i] = f2bf(tanhf(x[i] * cf[c] + cf[1024 + c]));
}

// BN2 + tanh + residual, permute rows (b,t,p) -> xs[t][b][p][h] (bf16) for the GRU
__global__ void k_bn2_apply(const float* __restrict__ y2, const float* __restrict__ cf,
                            const float* __restrict__ res, unsigned short* __restrict__ xs,
                            long long n) {
    long long i = (long long)blockIdx.x * blockDim.x + threadIdx.x;
    if (i >= n) return;
    int h = (int)(i & 1023);
    long long row = i >> 10;                  // (b*32+t)*12+p
    int p = (int)(row % 12);
    int t = (int)((row / 12) % 32);
    int b = (int)(row / 384);
    float v = tanhf(y2[i] * cf[h] + cf[1024 + h]) + res[i];
    xs[(((long long)t * 16 + b) * 12 + p) * 1024 + h] = f2bf(v);
}

// ---------------- im2col (temporal 3-tap, pad 1) + weight repack ----------------
__global__ void k_im2col(const unsigned short* __restrict__ y, unsigned short* __restrict__ a2,
                         long long n) {
    long long i = (long long)blockIdx.x * blockDim.x + threadIdx.x;
    if (i >= n) return;
    int col = (int)(i % 3072);
    long long row = i / 3072;                 // (b*32+t)*12+p
    int dt = col >> 10, h = col & 1023;
    int t = (int)((row / 12) % 32);
    int tt = t + dt - 1;
    unsigned short v = 0;
    if (tt >= 0 && tt < 32) v = y[(row + (long long)(dt - 1) * 12) * 1024 + h];
    a2[i] = v;
}

__global__ void k_repack_wt(const float* __restrict__ wt, unsigned short* __restrict__ wp,
                            long long n) {                 // n = 1024*3072
    long long i = (long long)blockIdx.x * blockDim.x + threadIdx.x;
    if (i >= n) return;
    int o = (int)(i / 3072);
    int rem = (int)(i % 3072);
    int dt = rem >> 10, h = rem & 1023;
    wp[i] = f2bf(wt[((long long)o * 1024 + h) * 3 + dt]);  // Wt[o][h][dt][0]
}

// ---------------- fused GRU gate ----------------
__global__ void k_gru_gate(const float* __restrict__ gi, const float* __restrict__ gh,
                           float* __restrict__ hp, unsigned short* __restrict__ hpb,
                           float* __restrict__ total, unsigned short* __restrict__ totalb, int t) {
    int i = blockIdx.x * 256 + threadIdx.x;   // 192*1024
    if (i >= 192 * 1024) return;
    int h = i & 1023, row = i >> 10;          // row = b*12+p
    long long gro = (long long)row * 3072;
    float ir = gi[gro + h], iz = gi[gro + 1024 + h], ig = gi[gro + 2048 + h];
    float hr = gh[gro + h], hz = gh[gro + 1024 + h], hg = gh[gro + 2048 + h];
    float r = 1.0f / (1.0f + expf(-(ir + hr)));
    float z = 1.0f / (1.0f + expf(-(iz + hz)));
    float nn = tanhf(ig + r * hg);
    float hprev = hp[i];
    float hn = (1.0f - z) * nn + z * hprev;
    hp[i] = hn;
    hpb[i] = f2bf(hn);
    int b = row / 12, p = row % 12;
    long long grow = ((long long)(b * 32 + t) * 12 + p) * 1024 + h;
    total[grow] = hn;
    totalb[grow] = f2bf(hn);
}

// ---------------- max over persons ----------------
__global__ void k_maxpool(const float* __restrict__ total, unsigned short* __restrict__ tm) {
    int i = blockIdx.x * 256 + threadIdx.x;   // 512*1024
    if (i >= 512 * 1024) return;
    int h = i & 1023, bt = i >> 10;           // bt = b*32+t
    float m = -3.4e38f;
    for (int p = 0; p < 12; ++p)
        m = fmaxf(m, total[((long long)bt * 12 + p) * 1024 + h]);
    tm[i] = f2bf(m);
}

// ---------------- host side ----------------
static void gemm(hipStream_t s,
                 const unsigned short* A, long long sA, int lda,
                 const unsigned short* Bm, long long sB, int ldb, int bWT,
                 const float* bias,
                 float* C, long long sC, unsigned short* Cb, long long sCb,
                 int M, int N, int K, int act, int ldc, int batch) {
    dim3 g((N + TBN - 1) / TBN, (M + TBM - 1) / TBM, batch);
    k_gemm<<<g, 256, 0, s>>>(A, sA, lda, Bm, sB, ldb, bWT, bias, C, sC, Cb, sCb, M, N, K, act, ldc);
}

static void cvt(hipStream_t s, const float* in, unsigned short* out, long long n) {
    k_cvt<<<(unsigned)((n + 255) / 256), 256, 0, s>>>(in, out, n);
}

extern "C" void kernel_launch(void* const* d_in, const int* in_sizes, int n_in,
                              void* d_out, int out_size, void* d_ws, size_t ws_size,
                              hipStream_t stream) {
    const int Bc = 16, T = 32, P = 12, F = 4096, E = 2048, H = 1024;
    const int N1 = Bc * T * P;        // 6144
    const int Nn = T * P;             // 384
    const int TH = 3 * H;             // 3072
    const int BP = Bc * P;            // 192
    const int BT = Bc * T;            // 512

    const float* feature = (const float*)d_in[0];
    const float* W1  = (const float*)d_in[1];  const float* b1  = (const float*)d_in[2];
    const float* W2  = (const float*)d_in[3];  const float* b2  = (const float*)d_in[4];
    const float* Wsp = (const float*)d_in[5];  const float* bsp = (const float*)d_in[6];
    const float* Wth = (const float*)d_in[7];  const float* bth = (const float*)d_in[8];
    const float* Wph = (const float*)d_in[9];  const float* bph = (const float*)d_in[10];
    const float* Wg  = (const float*)d_in[11];
    const float* g1  = (const float*)d_in[12]; const float* be1 = (const float*)d_in[13];
    const float* Wt  = (const float*)d_in[14]; const float* bt  = (const float*)d_in[15];
    const float* g2  = (const float*)d_in[16]; const float* be2 = (const float*)d_in[17];
    const float* Wih = (const float*)d_in[18]; const float* bih = (const float*)d_in[19];
    const float* Whh = (const float*)d_in[20]; const float* bhh = (const float*)d_in[21];
    const float* Wa  = (const float*)d_in[22]; const float* ba  = (const float*)d_in[23];
    const float* Wac = (const float*)d_in[24]; const float* bac = (const float*)d_in[25];

    // ---- workspace bump allocator (256B aligned) with phase aliasing ----
    size_t off = 0;
    auto alc = [&](size_t bytes) -> void* {
        off = (off + 255) & ~(size_t)255;
        void* p = (char*)d_ws + off;
        off += bytes;
        return p;
    };
    unsigned short* feat_bf = (unsigned short*)alc((size_t)N1 * F * 2);   // later: a2 (im2col)
    unsigned short* w1_bf   = (unsigned short*)alc((size_t)E * F * 2);    // later: total_bf
    unsigned short* h_bf    = (unsigned short*)alc((size_t)N1 * E * 2);   // later: spg_f
    unsigned short* w2_bf   = (unsigned short*)alc((size_t)H * E * 2);    // later: gi_f
    float*          x_f     = (float*)alc((size_t)N1 * H * 4);            // residual
    unsigned short* x_bf    = (unsigned short*)alc((size_t)N1 * H * 2);   // later: xs_bf
    unsigned short* wsp_bf  = (unsigned short*)alc((size_t)H * H * 2);
    unsigned short* wth_bf  = (unsigned short*)alc((size_t)H * H * 2);
    unsigned short* wph_bf  = (unsigned short*)alc((size_t)H * H * 2);
    unsigned short* wg_bf   = (unsigned short*)alc((size_t)H * H * 2);
    unsigned short* sp_bf   = (unsigned short*)alc((size_t)N1 * H * 2);   // sp+hs later: y2_f
    unsigned short* hs_bf   = (unsigned short*)alc((size_t)N1 * H * 2);
    unsigned short* ht_bf   = (unsigned short*)alc((size_t)N1 * H * 2);   // ht+v later: total_f
    unsigned short* v_bf    = (unsigned short*)alc((size_t)N1 * H * 2);
    float*          f_f     = (float*)alc((size_t)Bc * Nn * Nn * 4);      // later: gh/hprev
    unsigned short* ct_bf   = (unsigned short*)alc((size_t)Bc * Nn * Nn * 2); // later: tm
    unsigned short* y_bf    = (unsigned short*)alc((size_t)N1 * H * 2);
    unsigned short* wt_bf   = (unsigned short*)alc((size_t)H * TH * 2);
    unsigned short* wih_bf  = (unsigned short*)alc((size_t)TH * H * 2);
    unsigned short* whh_bf  = (unsigned short*)alc((size_t)TH * H * 2);
    float*          stats   = (float*)alc(2048 * 4);
    float*          coef    = (float*)alc(2048 * 4);

    // aliased phase-2 buffers
    unsigned short* a2_bf    = feat_bf;                                   // [N1,3072]
    unsigned short* total_bf = w1_bf;                                     // [N1,H]
    float*          spg_f    = (float*)h_bf;                              // [N1,H]
    float*          gi_f     = (float*)w2_bf;                             // [192,3072]
    unsigned short* xs_bf    = x_bf;                                      // [T,B,P,H]
    float*          y2_f     = (float*)sp_bf;                             // [N1,H] (spans sp+hs)
    float*          total_f  = (float*)ht_bf;                             // [N1,H] (spans ht+v)
    float*          gh_f     = f_f;                                       // [192,3072]
    float*          hprev_f  = (float*)((char*)f_f + (size_t)BP * TH * 4);
    unsigned short* hprev_bf = (unsigned short*)((char*)hprev_f + (size_t)BP * H * 4);
    unsigned short* tm_bf    = ct_bf;                                     // [512,H]

    float* out_action   = (float*)d_out;                 // [6144,9]
    float* out_activity = out_action + (size_t)N1 * 9;   // [512,8]

    // ---- 0) f32 -> bf16 conversions ----
    cvt(stream, feature, feat_bf, (long long)N1 * F);
    cvt(stream, W1,  w1_bf,  (long long)E * F);
    cvt(stream, W2,  w2_bf,  (long long)H * E);
    cvt(stream, Wsp, wsp_bf, (long long)H * H);
    cvt(stream, Wth, wth_bf, (long long)H * H);
    cvt(stream, Wph, wph_bf, (long long)H * H);
    cvt(stream, Wg,  wg_bf,  (long long)H * H);
    cvt(stream, Wih, wih_bf, (long long)TH * H);
    cvt(stream, Whh, whh_bf, (long long)TH * H);

    // ---- 1) embed: relu(feat @ W1^T + b1) -> tanh(h @ W2^T + b2) ----
    gemm(stream, feat_bf, 0, F, w1_bf, 0, F, 1, b1, nullptr, 0, h_bf, 0, N1, E, F, 1, E, 1);
    gemm(stream, h_bf, 0, E, w2_bf, 0, E, 1, b2, x_f, 0, x_bf, 0, N1, H, E, 2, H, 1);
    // (w1_bf, h_bf, w2_bf free from here -> reused)

    // ---- 2) 1x1 convs: sp/hs/ht rows [N1,H] ----
    gemm(stream, x_bf, 0, H, wsp_bf, 0, H, 1, bsp, nullptr, 0, sp_bf, 0, N1, H, H, 0, H, 1);
    gemm(stream, x_bf, 0, H, wth_bf, 0, H, 1, bth, nullptr, 0, hs_bf, 0, N1, H, H, 0, H, 1);
    gemm(stream, x_bf, 0, H, wph_bf, 0, H, 1, bph, nullptr, 0, ht_bf, 0, N1, H, H, 0, H, 1);

    // ---- 3) v = ht @ Wg (NN)  then batched f_b = v_b @ hs_b^T ----
    gemm(stream, ht_bf, 0, H, wg_bf, 0, H, 0, nullptr, nullptr, 0, v_bf, 0, N1, H, H, 0, H, 1);
    gemm(stream, v_bf, (long long)Nn * H, H, hs_bf, (long long)Nn * H, H, 1, nullptr,
         f_f, (long long)Nn * Nn, nullptr, 0, Nn, Nn, H, 0, Nn, Bc);

    // ---- 4) column softmax, write C^T (bf16) ----
    k_softmax_ct<<<(Bc * Nn + 255) / 256, 256, 0, stream>>>(f_f, ct_bf);

    // ---- 5) batched spg_b = C_b^T @ sp_b (NN) ----
    gemm(stream, ct_bf, (long long)Nn * Nn, Nn, sp_bf, (long long)Nn * H, H, 0, nullptr,
         spg_f, (long long)Nn * H, nullptr, 0, Nn, H, Nn, 0, H, Bc);

    // ---- 6) BN1 + tanh -> y_bf ----
    k_zero<<<8, 256, 0, stream>>>(stats, 2048);
    k_bn_stats<<<N1 / 64, 256, 0, stream>>>(spg_f, stats, N1);
    k_bn_coeff<<<4, 256, 0, stream>>>(stats, g1, be1, coef, 1.0f / (float)N1);
    k_bn1_apply<<<(unsigned)(((long long)N1 * H + 255) / 256), 256, 0, stream>>>(
        spg_f, coef, y_bf, (long long)N1 * H);

    // ---- 7) temporal conv via im2col (K=3H) ----
    k_im2col<<<(unsigned)(((long long)N1 * TH + 255) / 256), 256, 0, stream>>>(
        y_bf, a2_bf, (long long)N1 * TH);
    k_repack_wt<<<(unsigned)(((long long)H * TH + 255) / 256), 256, 0, stream>>>(
        Wt, wt_bf, (long long)H * TH);
    gemm(stream, a2_bf, 0, TH, wt_bf, 0, TH, 1, bt, y2_f, 0, nullptr, 0, N1, H, TH, 0, H, 1);

    // ---- 8) BN2 + tanh + residual -> xs[T,B,P,H] bf16 ----
    k_zero<<<8, 256, 0, stream>>>(stats, 2048);
    k_bn_stats<<<N1 / 64, 256, 0, stream>>>(y2_f, stats, N1);
    k_bn_coeff<<<4, 256, 0, stream>>>(stats, g2, be2, coef, 1.0f / (float)N1);
    k_bn2_apply<<<(unsigned)(((long long)N1 * H + 255) / 256), 256, 0, stream>>>(
        y2_f, coef, x_f, xs_bf, (long long)N1 * H);

    // ---- 9) GRU scan over T ----
    k_zero<<<(BP * H + 255) / 256, 256, 0, stream>>>(hprev_f, (long long)BP * H);
    k_zero<<<(BP * H / 2 + 255) / 256, 256, 0, stream>>>((float*)hprev_bf, (long long)BP * H / 2);
    for (int t = 0; t < T; ++t) {
        const unsigned short* xt = xs_bf + (long long)t * BP * H;
        gemm(stream, xt, 0, H, wih_bf, 0, H, 1, bih, gi_f, 0, nullptr, 0, BP, TH, H, 0, TH, 1);
        gemm(stream, hprev_bf, 0, H, whh_bf, 0, H, 1, bhh, gh_f, 0, nullptr, 0, BP, TH, H, 0, TH, 1);
        k_gru_gate<<<(BP * H + 255) / 256, 256, 0, stream>>>(
            gi_f, gh_f, hprev_f, hprev_bf, total_f, total_bf, t);
    }

    // ---- 10) heads ----
    cvt(stream, Wa,  wt_bf, 9 * H);                  // reuse wt_bf for head weights
    cvt(stream, Wac, wt_bf + 16 * H, 8 * H);
    gemm(stream, total_bf, 0, H, wt_bf, 0, H, 1, ba, out_action, 0, nullptr, 0,
         N1, 9, H, 0, 9, 1);
    k_maxpool<<<(BT * H + 255) / 256, 256, 0, stream>>>(total_f, tm_bf);
    gemm(stream, tm_bf, 0, H, wt_bf + 16 * H, 0, H, 1, bac, out_activity, 0, nullptr, 0,
         BT, 8, H, 0, 8, 1);
}
